// MultiHeadAttention_35150012351331
// MI455X (gfx1250) — compile-verified, matrix operand order checked by reading, not run on hardware
//
#include <hip/hip_runtime.h>
#include <hip/hip_bf16.h>

// MultiHeadAttention for MI455X (gfx1250), wave32 + v_wmma_f32_16x16x32_f16.
// Pipeline:
//   0) one-shot fp32->fp16 conversion of weights (8 MiB) and q/k/v (48 MiB)
//   1) 3x QKV projection GEMMs (pure fp16 WMMA, V stored transposed)
//   2) causal flash attention (online softmax, P staged through LDS)
//   3) output projection GEMM (fp16 x fp16 -> fp32, coalesced stores)
// Requires ws_size >= 104 MiB (oc aliases the dead fp16-q buffer).

#define D_MODEL  1024
#define N_HEAD   16
#define HEAD_DIM 64
#define BATCH    4
#define SEQ      2048
#define ROWS     (BATCH * SEQ)   // 8192

typedef __attribute__((ext_vector_type(16))) _Float16 v16h;
typedef __attribute__((ext_vector_type(8)))  _Float16 v8h;
typedef __attribute__((ext_vector_type(4)))  _Float16 v4h;
typedef __attribute__((ext_vector_type(8)))  float    v8f;
typedef __attribute__((ext_vector_type(4)))  float    v4f;

// ---------------------------------------------------------------------------
// Fragment loader.
// 16-bit A-matrix 16x32 layout (ISA 7.12.2): lane&15 = row (A) / col (B),
// lanes 0-15 hold K={k0..k0+7, k0+16..k0+23}, lanes 16-31 hold K offset +8.
// B operand is layout-symmetric with lane&15 = N column; both our GEMM
// operands are contraction-contiguous row-major, so one gather serves both.
// ---------------------------------------------------------------------------
__device__ __forceinline__ v16h frag_from_f16(const _Float16* __restrict__ base,
                                              int row0, long ld, int k0, int lane) {
  const _Float16* p = base + (long)(row0 + (lane & 15)) * ld + k0 + ((lane >> 4) << 3);
  v8h lo = *(const v8h*)(p);
  v8h hi = *(const v8h*)(p + 16);
  v16h r;
#pragma unroll
  for (int i = 0; i < 8; ++i) { r[i] = lo[i]; r[8 + i] = hi[i]; }
  return r;
}

__device__ __forceinline__ v8f wmma16(v16h a, v16h b, v8f c) {
  // D = A(16x32) * B(32x16) + C(16x16), f32 accumulate
  return __builtin_amdgcn_wmma_f32_16x16x32_f16(false, a, false, b, (short)0, c,
                                                false, false);
}

// ---------------------------------------------------------------------------
// Kernel 0: fp32 -> fp16 conversion, 4 elements per thread.
// ---------------------------------------------------------------------------
__global__ __launch_bounds__(256)
void cvt_f16_kernel(const float* __restrict__ X, _Float16* __restrict__ Xh) {
  int i = (blockIdx.x * 256 + threadIdx.x) * 4;
  v4f x = *(const v4f*)(X + i);
  v4h y;
#pragma unroll
  for (int j = 0; j < 4; ++j) y[j] = (_Float16)x[j];
  *(v4h*)(Xh + i) = y;
}

// ---------------------------------------------------------------------------
// Kernel 1: Y = X @ W^T + bias, all-fp16 operands, fp32 accumulate.
// mode 0: Q -> [b,h,s,64] scaled by 1/sqrt(64)
// mode 1: K -> [b,h,s,64]
// mode 2: V -> [b,h,64,s]  (transposed so PV B-fragments are contiguous)
// Block = 8 waves; wave owns a 32x64 tile (2 A frags x 4 shared B frags,
// 8 WMMAs per K step). Block covers 256 rows x 64 cols.
// ---------------------------------------------------------------------------
__global__ __launch_bounds__(256)
void proj_qkv_kernel(const _Float16* __restrict__ Xh, const _Float16* __restrict__ Wh,
                     const float* __restrict__ bias, _Float16* __restrict__ out,
                     int mode, float scale) {
  const int lane  = threadIdx.x & 31;
  const int wave  = threadIdx.x >> 5;
  const int mbase = blockIdx.x * 256 + wave * 32;
  const int nbase = blockIdx.y * 64;

  v8f acc[2][4];
#pragma unroll
  for (int j = 0; j < 4; ++j) {
    float bv = bias[nbase + j * 16 + (lane & 15)];   // bias per output column
#pragma unroll
    for (int r = 0; r < 8; ++r) { acc[0][j][r] = bv; acc[1][j][r] = bv; }
  }

  const _Float16* arow = Xh + (long)(mbase + (lane & 15)) * D_MODEL;
  for (int k = 0; k < D_MODEL; k += 32) {
    // prefetch next K-step of this wave's A rows (global_prefetch_b8)
    __builtin_prefetch(arow + k + 32, 0, 3);
    v16h a0 = frag_from_f16(Xh, mbase,      D_MODEL, k, lane);
    v16h a1 = frag_from_f16(Xh, mbase + 16, D_MODEL, k, lane);
#pragma unroll
    for (int j = 0; j < 4; ++j) {
      v16h b = frag_from_f16(Wh, nbase + j * 16, D_MODEL, k, lane); // W row n = B col n
      acc[0][j] = wmma16(a0, b, acc[0][j]);
      acc[1][j] = wmma16(a1, b, acc[1][j]);
    }
  }

  // C layout: VGPR r, lane l -> (m = r + 8*(l>=16), n = l&15)
#pragma unroll
  for (int mi = 0; mi < 2; ++mi) {
#pragma unroll
    for (int j = 0; j < 4; ++j) {
      int n = nbase + j * 16 + (lane & 15);
      int h = n >> 6, d = n & 63;
#pragma unroll
      for (int r = 0; r < 8; ++r) {
        int m = mbase + mi * 16 + r + ((lane >> 4) << 3);
        int b = m >> 11, s = m & (SEQ - 1);
        _Float16 val = (_Float16)(acc[mi][j][r] * scale);
        long idx;
        if (mode == 2) idx = ((long)(b * N_HEAD + h) * HEAD_DIM + d) * SEQ + s;
        else           idx = ((long)(b * N_HEAD + h) * SEQ + s) * HEAD_DIM + d;
        out[idx] = val;
      }
    }
  }
}

// ---------------------------------------------------------------------------
// Kernel 2: causal flash attention. One wave per 16-row q tile.
// Scores S = (Q*scale) K^T via WMMA; online softmax with half-wave shuffle
// row reductions (C-layout holds each row in one VGPR across 16 lanes);
// P goes through LDS for C-layout -> A-layout; PV via WMMA vs transposed V.
// ---------------------------------------------------------------------------
__global__ __launch_bounds__(256)
void attn_kernel(const _Float16* __restrict__ Qh, const _Float16* __restrict__ Kh,
                 const _Float16* __restrict__ Vt, _Float16* __restrict__ Oc) {
  __shared__ _Float16 plds[8][16 * 32];   // per-wave 16x32 P staging (8 KiB)

  const int lane  = threadIdx.x & 31;
  const int wave  = threadIdx.x >> 5;
  const int qt    = blockIdx.x * 8 + wave;   // q tile 0..127
  const int bh    = blockIdx.y;              // b*16 + h
  const int qbase = qt * 16;

  const _Float16* q  = Qh + (long)bh * SEQ * HEAD_DIM;
  const _Float16* kk = Kh + (long)bh * SEQ * HEAD_DIM;
  const _Float16* vt = Vt + (long)bh * HEAD_DIM * SEQ;
  _Float16* ph = plds[wave];

  // Q fragments: 16 rows x 64 contraction = two K=32 A-fragments (pre-scaled)
  v16h qa0 = frag_from_f16(q, qbase, HEAD_DIM, 0,  lane);
  v16h qa1 = frag_from_f16(q, qbase, HEAD_DIM, 32, lane);

  v8f o[4];
#pragma unroll
  for (int j = 0; j < 4; ++j)
#pragma unroll
    for (int r = 0; r < 8; ++r) o[j][r] = 0.0f;

  float mrow[8], lrow[8];
#pragma unroll
  for (int r = 0; r < 8; ++r) { mrow[r] = -1e30f; lrow[r] = 0.0f; }

  const int rowb   = qbase + ((lane >> 4) << 3);  // row of element r is rowb + r
  const int nchunk = (qt >> 1) + 1;               // 32-wide kv chunks (causal)

  for (int c = 0; c < nchunk; ++c) {
    const int kvb = c * 32;

    // ---- scores: two 16x16 tiles, K=64 contraction -------------------------
    v8f s0 = {}; v8f s1 = {};
    {
      v16h kb;
      kb = frag_from_f16(kk, kvb,      HEAD_DIM, 0,  lane); s0 = wmma16(qa0, kb, s0);
      kb = frag_from_f16(kk, kvb,      HEAD_DIM, 32, lane); s0 = wmma16(qa1, kb, s0);
      kb = frag_from_f16(kk, kvb + 16, HEAD_DIM, 0,  lane); s1 = wmma16(qa0, kb, s1);
      kb = frag_from_f16(kk, kvb + 16, HEAD_DIM, 32, lane); s1 = wmma16(qa1, kb, s1);
    }

    // ---- causal mask (only the last chunk touches the diagonal) ------------
    if (c == nchunk - 1) {
      const int col0 = kvb + (lane & 15);
      const int col1 = col0 + 16;
#pragma unroll
      for (int r = 0; r < 8; ++r) {
        if (col0 > rowb + r) s0[r] = -1e30f;
        if (col1 > rowb + r) s1[r] = -1e30f;
      }
    }

    // ---- online softmax -----------------------------------------------------
#pragma unroll
    for (int r = 0; r < 8; ++r) {
      float sm = fmaxf(s0[r], s1[r]);
#pragma unroll
      for (int off = 1; off < 16; off <<= 1) sm = fmaxf(sm, __shfl_xor(sm, off, 32));
      float mnew  = fmaxf(mrow[r], sm);
      float alpha = __expf(mrow[r] - mnew);
      mrow[r] = mnew;
      float p0 = __expf(s0[r] - mnew);
      float p1 = __expf(s1[r] - mnew);
      float ps = p0 + p1;
#pragma unroll
      for (int off = 1; off < 16; off <<= 1) ps += __shfl_xor(ps, off, 32);
      lrow[r] = lrow[r] * alpha + ps;
#pragma unroll
      for (int j = 0; j < 4; ++j) o[j][r] *= alpha;
      // stage P (C layout -> row-major 16x32 in LDS)
      int m = r + ((lane >> 4) << 3);
      int n = lane & 15;
      ph[m * 32 + n]      = (_Float16)p0;
      ph[m * 32 + 16 + n] = (_Float16)p1;
    }

    // ---- PV: P(16x32) x V(32x64) using transposed-V contiguous B frags -----
    v16h pa = frag_from_f16(ph, 0, 32, 0, lane);
#pragma unroll
    for (int j = 0; j < 4; ++j) {
      v16h vb = frag_from_f16(vt, j * 16, SEQ, kvb, lane);
      o[j] = wmma16(pa, vb, o[j]);
    }
  }

  // ---- normalize and store concat-head layout [b, s, h*64+d] as fp16 -------
  const int b = bh >> 4, h = bh & 15;
#pragma unroll
  for (int r = 0; r < 8; ++r) {
    float inv = 1.0f / lrow[r];
    int   m   = rowb + r;
    long  base = (long)(b * SEQ + m) * D_MODEL + h * HEAD_DIM;
#pragma unroll
    for (int j = 0; j < 4; ++j)
      Oc[base + j * 16 + (lane & 15)] = (_Float16)(o[j][r] * inv);
  }
}

// ---------------------------------------------------------------------------
// Kernel 3: Y = Oc @ Wo^T + bo, fp16 x fp16 -> fp32 output (coalesced).
// Same 32x64-per-wave tiling as the QKV projections.
// ---------------------------------------------------------------------------
__global__ __launch_bounds__(256)
void proj_out_kernel(const _Float16* __restrict__ Xh, const _Float16* __restrict__ Wh,
                     const float* __restrict__ bias, float* __restrict__ Y) {
  const int lane  = threadIdx.x & 31;
  const int wave  = threadIdx.x >> 5;
  const int mbase = blockIdx.x * 256 + wave * 32;
  const int nbase = blockIdx.y * 64;

  v8f acc[2][4];
#pragma unroll
  for (int j = 0; j < 4; ++j) {
    float bv = bias[nbase + j * 16 + (lane & 15)];
#pragma unroll
    for (int r = 0; r < 8; ++r) { acc[0][j][r] = bv; acc[1][j][r] = bv; }
  }

  const _Float16* arow = Xh + (long)(mbase + (lane & 15)) * D_MODEL;
  for (int k = 0; k < D_MODEL; k += 32) {
    __builtin_prefetch(arow + k + 32, 0, 3);
    v16h a0 = frag_from_f16(Xh, mbase,      D_MODEL, k, lane);
    v16h a1 = frag_from_f16(Xh, mbase + 16, D_MODEL, k, lane);
#pragma unroll
    for (int j = 0; j < 4; ++j) {
      v16h b = frag_from_f16(Wh, nbase + j * 16, D_MODEL, k, lane);
      acc[0][j] = wmma16(a0, b, acc[0][j]);
      acc[1][j] = wmma16(a1, b, acc[1][j]);
    }
  }

#pragma unroll
  for (int mi = 0; mi < 2; ++mi) {
#pragma unroll
    for (int r = 0; r < 8; ++r) {
      long m = mbase + mi * 16 + r + ((lane >> 4) << 3);
#pragma unroll
      for (int j = 0; j < 4; ++j)
        Y[m * D_MODEL + nbase + j * 16 + (lane & 15)] = acc[mi][j][r];
    }
  }
}

// ---------------------------------------------------------------------------
extern "C" void kernel_launch(void* const* d_in, const int* in_sizes, int n_in,
                              void* d_out, int out_size, void* d_ws, size_t ws_size,
                              hipStream_t stream) {
  const float* q   = (const float*)d_in[0];
  const float* k   = (const float*)d_in[1];
  const float* v   = (const float*)d_in[2];
  // d_in[3] is the causal (tril) mask: applied analytically in attn_kernel.
  const float* w_q = (const float*)d_in[4];
  const float* b_q = (const float*)d_in[5];
  const float* w_k = (const float*)d_in[6];
  const float* b_k = (const float*)d_in[7];
  const float* w_v = (const float*)d_in[8];
  const float* b_v = (const float*)d_in[9];
  const float* w_o = (const float*)d_in[10];
  const float* b_o = (const float*)d_in[11];

  char* ws = (char*)d_ws;
  const size_t WBYTES = (size_t)D_MODEL * D_MODEL * sizeof(_Float16);               // 2 MiB
  const size_t XBYTES = (size_t)ROWS * D_MODEL * sizeof(_Float16);                  // 16 MiB
  const size_t HBYTES = (size_t)BATCH * N_HEAD * SEQ * HEAD_DIM * sizeof(_Float16); // 16 MiB
  _Float16* wq16 = (_Float16*)(ws);
  _Float16* wk16 = (_Float16*)(ws + 1 * WBYTES);
  _Float16* wv16 = (_Float16*)(ws + 2 * WBYTES);
  _Float16* wo16 = (_Float16*)(ws + 3 * WBYTES);
  char* xs = ws + 4 * WBYTES;
  _Float16* xq16 = (_Float16*)(xs);               // dead after Q projection
  _Float16* xk16 = (_Float16*)(xs + 1 * XBYTES);
  _Float16* xv16 = (_Float16*)(xs + 2 * XBYTES);
  char* act = xs + 3 * XBYTES;
  _Float16* qh = (_Float16*)(act);
  _Float16* kh = (_Float16*)(act + HBYTES);
  _Float16* vt = (_Float16*)(act + 2 * HBYTES);
  _Float16* oc = xq16;   // alias: xq16 is dead once the Q projection has run

  // 0) fp32 -> fp16 conversions (one-shot)
  const int cvtw = (D_MODEL * D_MODEL / 4) / 256;   // 1024 blocks
  const int cvtx = (ROWS * D_MODEL / 4) / 256;      // 8192 blocks
  cvt_f16_kernel<<<cvtw, 256, 0, stream>>>(w_q, wq16);
  cvt_f16_kernel<<<cvtw, 256, 0, stream>>>(w_k, wk16);
  cvt_f16_kernel<<<cvtw, 256, 0, stream>>>(w_v, wv16);
  cvt_f16_kernel<<<cvtw, 256, 0, stream>>>(w_o, wo16);
  cvt_f16_kernel<<<cvtx, 256, 0, stream>>>(q, xq16);
  cvt_f16_kernel<<<cvtx, 256, 0, stream>>>(k, xk16);
  cvt_f16_kernel<<<cvtx, 256, 0, stream>>>(v, xv16);

  // 1) QKV projections (pure fp16 WMMA)
  dim3 gblk(ROWS / 256, D_MODEL / 64);  // (32, 16)
  proj_qkv_kernel<<<gblk, 256, 0, stream>>>(xq16, wq16, b_q, qh, 0, 0.125f); // 1/sqrt(64)
  proj_qkv_kernel<<<gblk, 256, 0, stream>>>(xk16, wk16, b_k, kh, 1, 1.0f);
  proj_qkv_kernel<<<gblk, 256, 0, stream>>>(xv16, wv16, b_v, vt, 2, 1.0f);

  // 2) causal flash attention (oc reuses the dead xq16 region)
  attn_kernel<<<dim3(SEQ / 16 / 8, BATCH * N_HEAD), 256, 0, stream>>>(qh, kh, vt, oc);

  // 3) output projection
  proj_out_kernel<<<gblk, 256, 0, stream>>>(oc, wo16, b_o, (float*)d_out);
}
